// AnotherAttentionMLP_45294725103667
// MI455X (gfx1250) — compile-verified, matrix operand order checked by reading, not run on hardware
//
#include <hip/hip_runtime.h>

// ---------------------------------------------------------------------------
// AnotherAttentionMLP for MI455X (gfx1250, wave32, WMMA).
// bf16 v_wmma_f32_16x16x32_bf16 GEMMs, fp32 accumulation.
// GEMM: block tile 128x128, per-wave 32x64 (8 WMMA / wave / K-step),
// double-buffered LDS staging via GLOBAL_LOAD_ASYNC_TO_LDS_B128 (ASYNCcnt).
// All GEMM dims are multiples of 128(M) x 128(N) x 32(K) -> no tails.
// ---------------------------------------------------------------------------

typedef __attribute__((ext_vector_type(16))) __bf16 v16bf;
typedef __attribute__((ext_vector_type(8)))  float  v8f;

#define TPB 256  // 8 waves of 32

// ---------------------------------------------------------------------------
// Weight pack: fp32 [K,N] row-major -> bf16 WMMA-fragment tile order.
// Tile (kt,nt) covers K rows [kt*32, kt*32+32), N cols [nt*16, nt*16+16).
// Within a tile: lane l owns 16 contiguous bf16 at offset l*16:
//   n = nt*16 + (l%16),  k = kt*32 + (l/16)*16 + j   (j = 0..15)
// Tiles stored nt-major:  tile_index = nt*(K/32) + kt.
// ---------------------------------------------------------------------------
__global__ __launch_bounds__(TPB)
void pack_w_kernel(const float* __restrict__ W, __bf16* __restrict__ P,
                   int K, int N) {
  const int idx = blockIdx.x * TPB + threadIdx.x;
  if (idx >= K * N) return;
  const int tile = idx >> 9;           // 512 elements per 32x16 tile
  const int rem  = idx & 511;
  const int l    = rem >> 4;           // lane 0..31
  const int j    = rem & 15;
  const int ktiles = K >> 5;
  const int nt = tile / ktiles;
  const int kt = tile - nt * ktiles;
  const int k  = (kt << 5) + ((l >> 4) << 4) + j;
  const int n  = (nt << 4) + (l & 15);
  P[idx] = (__bf16)W[(size_t)k * N + n];
}

// ---------------------------------------------------------------------------
// Row LayerNorm (fp32 in) -> bf16 out.  One 256-thread block per row.
// ---------------------------------------------------------------------------
__global__ __launch_bounds__(TPB)
void ln_to_bf16_kernel(const float* __restrict__ X,
                       const float* __restrict__ g,
                       const float* __restrict__ b,
                       __bf16* __restrict__ Y, int D) {
  const int row = blockIdx.x;
  const int tid = threadIdx.x;
  const float* x = X + (size_t)row * D;

  float s = 0.0f, ss = 0.0f;
  for (int i = tid; i < D; i += TPB) {
    const float v = x[i];
    s += v; ss += v * v;
  }
  #pragma unroll
  for (int off = 16; off > 0; off >>= 1) {
    s  += __shfl_xor(s,  off, 32);
    ss += __shfl_xor(ss, off, 32);
  }
  __shared__ float ps[8], pss[8];
  __shared__ float mu_s, rs_s;
  if ((tid & 31) == 0) { ps[tid >> 5] = s; pss[tid >> 5] = ss; }
  __syncthreads();
  if (tid == 0) {
    float S = 0.0f, SS = 0.0f;
    #pragma unroll
    for (int i = 0; i < 8; ++i) { S += ps[i]; SS += pss[i]; }
    const float mu  = S / (float)D;
    const float var = SS / (float)D - mu * mu;
    mu_s = mu;
    rs_s = rsqrtf(var + 1e-5f);
  }
  __syncthreads();
  const float mu = mu_s, rs = rs_s;
  __bf16* y = Y + (size_t)row * D;
  for (int i = tid; i < D; i += TPB)
    y[i] = (__bf16)((x[i] - mu) * rs * g[i] + b[i]);
}

// ---------------------------------------------------------------------------
// GEMM: out[M,N] = epilogue( A[M,K](bf16,row-major) @ Bpacked[K,N](bf16) )
//
// Block: 8 waves, tile 128(M) x 128(N).  Wave (wm = wave&3, wn = wave>>2)
// owns rows [wm*32, wm*32+32) x cols [wn*64, wn*64+64): 2x4 WMMA tiles.
// Per K-step (K32): A 128x32 tile and B 8 fragment-tiles (8KB each) are
// DMA'd into double-buffered LDS with GLOBAL_LOAD_ASYNC_TO_LDS_B128 and
// overlapped with WMMA on the other buffer (s_wait_asynccnt pipelining).
// Epilogue: +bias, optional ReLU, optional fp32 residual add, fp32|bf16 out.
// ---------------------------------------------------------------------------
__global__ __launch_bounds__(TPB)
void gemm_bf16_wmma_kernel(const __bf16* __restrict__ A,
                           const __bf16* __restrict__ Bp,
                           const float*  __restrict__ bias,
                           const float*  __restrict__ resid,
                           float*        __restrict__ outF,
                           __bf16*       __restrict__ outB,
                           int M, int N, int K, int relu) {
  __shared__ __align__(32) __bf16 sA[2][128 * 32];   // 2 x 8KB
  __shared__ __align__(32) __bf16 sB[2][8 * 512];    // 2 x 8KB (fragment order)

  const int tid  = threadIdx.x;
  const int wave = tid >> 5;
  const int lane = tid & 31;
  const int m0   = blockIdx.x * 128;
  const int n0   = blockIdx.y * 128;
  const int wm   = wave & 3;   // 0..3 -> 32-row slab
  const int wn   = wave >> 2;  // 0..1 -> 64-col slab
  const int ktiles = K >> 5;

  v8f acc[2][4] = {};

  // ---- staging assignments (32B per thread for A, 32B for B, per K-step) ----
  const int arow  = tid >> 1;          // 0..127
  const int ahalf = (tid & 1) << 4;    // 0 or 16 (bf16 elements)
  const __bf16* aG = A + (size_t)(m0 + arow) * K + ahalf;
  // B: wave w stages fragment-tile ntg = n0/16 + w; lane copies [lane*16, +16)
  const __bf16* bG =
      Bp + ((size_t)((n0 >> 4) + wave) * (size_t)ktiles) * 512 + (size_t)lane * 16;

  // LDS byte addresses for async DMA (generic-pointer low 32 bits == LDS addr)
  unsigned aL[2], bL[2];
  aL[0] = (unsigned)(size_t)&sA[0][arow * 32 + ahalf];
  aL[1] = (unsigned)(size_t)&sA[1][arow * 32 + ahalf];
  bL[0] = (unsigned)(size_t)&sB[0][wave * 512 + lane * 16];
  bL[1] = (unsigned)(size_t)&sB[1][wave * 512 + lane * 16];

  auto stage = [&](int kt, int buf) {
    const __bf16* ag = aG + ((size_t)kt << 5);   // +kt*32 elements
    const __bf16* bg = bG + ((size_t)kt << 9);   // +kt*512 elements
    asm volatile("global_load_async_to_lds_b128 %0, %1, off"
                 :: "v"(aL[buf]),       "v"(ag)     : "memory");
    asm volatile("global_load_async_to_lds_b128 %0, %1, off"
                 :: "v"(aL[buf] + 16u), "v"(ag + 8) : "memory");
    asm volatile("global_load_async_to_lds_b128 %0, %1, off"
                 :: "v"(bL[buf]),       "v"(bg)     : "memory");
    asm volatile("global_load_async_to_lds_b128 %0, %1, off"
                 :: "v"(bL[buf] + 16u), "v"(bg + 8) : "memory");
  };

  // fragment read offsets (constant across K loop)
  const int aOffLo = (wm * 32 +      (lane & 15)) * 32 + ((lane >> 4) << 4);
  const int aOffHi = (wm * 32 + 16 + (lane & 15)) * 32 + ((lane >> 4) << 4);
  const int bOffBase = (wn * 4) * 512 + lane * 16;

  stage(0, 0);
  for (int kt = 0; kt < ktiles; ++kt) {
    const int buf = kt & 1;
    if (kt + 1 < ktiles) {
      stage(kt + 1, buf ^ 1);
      // 4 newer ops in flight; wait for the 4 older (in-order) to land.
      asm volatile("s_wait_asynccnt 0x4" ::: "memory");
    } else {
      asm volatile("s_wait_asynccnt 0x0" ::: "memory");
    }
    __syncthreads();

    const v16bf af0 = *(const v16bf*)(&sA[buf][aOffLo]);
    const v16bf af1 = *(const v16bf*)(&sA[buf][aOffHi]);
    #pragma unroll
    for (int s = 0; s < 4; ++s) {
      const v16bf bfr = *(const v16bf*)(&sB[buf][bOffBase + s * 512]);
      acc[0][s] = __builtin_amdgcn_wmma_f32_16x16x32_bf16(
          false, af0, false, bfr, (short)0, acc[0][s], false, false);
      acc[1][s] = __builtin_amdgcn_wmma_f32_16x16x32_bf16(
          false, af1, false, bfr, (short)0, acc[1][s], false, false);
    }
    __syncthreads();  // protect buffer reuse before next stage overwrites
  }

  // ---- epilogue ----
  // C/D layout: VGPR v, lanes 0-15 -> M = v; lanes 16-31 -> M = v+8; N = lane%16.
  #pragma unroll
  for (int i = 0; i < 2; ++i) {
    #pragma unroll
    for (int s = 0; s < 4; ++s) {
      const int n = n0 + (wn * 4 + s) * 16 + (lane & 15);
      const float bn = bias[n];
      #pragma unroll
      for (int v = 0; v < 8; ++v) {
        const int m = m0 + wm * 32 + i * 16 + ((lane >> 4) << 3) + v;
        float val = acc[i][s][v] + bn;
        if (relu)  val = fmaxf(val, 0.0f);
        if (resid) val += resid[(size_t)m * N + n];
        if (outB)  outB[(size_t)m * N + n] = (__bf16)val;
        else       outF[(size_t)m * N + n] = val;
      }
    }
  }
}

// ---------------------------------------------------------------------------
// Per-sample time-mean + transpose: r[b,o,t] = bf16(fc[b,t,o] + mean_t fc[b,:,o])
// ---------------------------------------------------------------------------
__global__ __launch_bounds__(TPB)
void mean_transpose_kernel(const float* __restrict__ fc,
                           __bf16* __restrict__ r) {
  const int bsi = blockIdx.x;   // 0..63
  const int tid = threadIdx.x;
  for (int o = tid; o < 512; o += TPB) {
    const float* col = fc + (size_t)bsi * 256 * 512 + o;
    float s = 0.0f;
    for (int t = 0; t < 256; ++t) s += col[(size_t)t * 512];
    const float mean = s * (1.0f / 256.0f);
    __bf16* dst = r + ((size_t)bsi * 512 + o) * 256;
    for (int t = 0; t < 256; ++t)
      dst[t] = (__bf16)(col[(size_t)t * 512] + mean);
  }
}

// ---------------------------------------------------------------------------
// Fused softmax over time (256) + weighted sum:
//   out[b,o] = sum_t fc[b,t,o] * softmax_t(w[b,o,:])[t]
// One wave32 per (b,o) row; 8 rows per block.
// ---------------------------------------------------------------------------
__global__ __launch_bounds__(TPB)
void softmax_wsum_kernel(const float* __restrict__ w,
                         const float* __restrict__ fc,
                         float* __restrict__ out) {
  const int tid  = threadIdx.x;
  const int wave = tid >> 5;
  const int lane = tid & 31;
  const int m    = blockIdx.x * 8 + wave;   // 0..32767
  const int bsi  = m >> 9;
  const int o    = m & 511;

  const float* wr = w + (size_t)m * 256;
  float vals[8];
  float mx = -3.0e38f;
  #pragma unroll
  for (int j = 0; j < 8; ++j) {
    vals[j] = wr[lane + j * 32];
    mx = fmaxf(mx, vals[j]);
  }
  #pragma unroll
  for (int off = 16; off > 0; off >>= 1)
    mx = fmaxf(mx, __shfl_xor(mx, off, 32));

  float sum = 0.0f;
  #pragma unroll
  for (int j = 0; j < 8; ++j) {
    vals[j] = __expf(vals[j] - mx);
    sum += vals[j];
  }
  #pragma unroll
  for (int off = 16; off > 0; off >>= 1)
    sum += __shfl_xor(sum, off, 32);

  float acc = 0.0f;
  const float* col = fc + (size_t)bsi * 256 * 512 + o;
  #pragma unroll
  for (int j = 0; j < 8; ++j)
    acc += vals[j] * col[(size_t)(lane + j * 32) * 512];
  #pragma unroll
  for (int off = 16; off > 0; off >>= 1)
    acc += __shfl_xor(acc, off, 32);

  if (lane == 0) out[m] = acc / sum;
}

// ---------------------------------------------------------------------------
// Host orchestration
// ---------------------------------------------------------------------------
extern "C" void kernel_launch(void* const* d_in, const int* in_sizes, int n_in,
                              void* d_out, int out_size, void* d_ws, size_t ws_size,
                              hipStream_t stream) {
  (void)in_sizes; (void)n_in; (void)out_size; (void)ws_size;

  constexpr int BSZ = 64, LEN = 256, FS = 512, H = 1024, OUTD = 512, NL = 4;
  constexpr int M1 = BSZ * LEN;   // 16384 tokens
  constexpr int M2 = BSZ * OUTD;  // 32768 attn rows
  constexpr int H3 = 3 * H;       // 3072

  const float* x     = (const float*)d_in[0];
  const float* ln0_g = (const float*)d_in[1];
  const float* ln0_b = (const float*)d_in[2];
  const float* W0    = (const float*)d_in[3];
  const float* b0    = (const float*)d_in[4];
  const float* rln_g = (const float*)d_in[5];
  const float* rln_b = (const float*)d_in[6];
  const float* rW    = (const float*)d_in[7];
  const float* rb    = (const float*)d_in[8];
  const float* lnf_g = (const float*)d_in[9];
  const float* lnf_b = (const float*)d_in[10];
  const float* Wf    = (const float*)d_in[11];
  const float* bfb   = (const float*)d_in[12];
  const float* Wa1   = (const float*)d_in[13];
  const float* ba1   = (const float*)d_in[14];
  const float* Wa2   = (const float*)d_in[15];
  const float* ba2   = (const float*)d_in[16];
  float* outp = (float*)d_out;

  // ---- workspace carve (256B aligned chunks) ----
  char* ws = (char*)d_ws;
  auto carve = [&](size_t bytes) -> char* {
    char* p = ws;
    ws += (bytes + 255) & ~(size_t)255;
    return p;
  };
  __bf16* W0p  = (__bf16*)carve((size_t)FS * H * 2);
  __bf16* rWp  = (__bf16*)carve((size_t)NL * H * H * 2);
  __bf16* Wfp  = (__bf16*)carve((size_t)H * OUTD * 2);
  __bf16* Wa1p = (__bf16*)carve((size_t)LEN * H3 * 2);
  __bf16* Wa2p = (__bf16*)carve((size_t)H3 * LEN * 2);
  float*  hA   = (float*) carve((size_t)M1 * H * 4);
  float*  hB   = (float*) carve((size_t)M1 * H * 4);
  __bf16* xh   = (__bf16*)carve((size_t)M1 * H * 2);   // LN output staging
  float*  fc   = (float*) carve((size_t)M1 * OUTD * 4);
  __bf16* rbuf = (__bf16*)carve((size_t)M2 * LEN * 2);
  __bf16* a1   = (__bf16*)carve((size_t)M2 * H3 * 2);
  float*  wbuf = (float*) carve((size_t)M2 * LEN * 4);

  auto pack = [&](const float* W, __bf16* P, int K, int N) {
    const int total = K * N;
    pack_w_kernel<<<(total + TPB - 1) / TPB, TPB, 0, stream>>>(W, P, K, N);
  };
  auto gemm = [&](const __bf16* A, const __bf16* Bp, const float* bias,
                  const float* resid, float* oF, __bf16* oB,
                  int M, int N, int K, int relu) {
    dim3 grid(M / 128, N / 128);
    gemm_bf16_wmma_kernel<<<grid, TPB, 0, stream>>>(A, Bp, bias, resid,
                                                    oF, oB, M, N, K, relu);
  };

  // ---- weight conversion + packing (bf16, fragment order) ----
  pack(W0, W0p, FS, H);
  for (int i = 0; i < NL; ++i)
    pack(rW + (size_t)i * H * H, rWp + (size_t)i * H * H, H, H);
  pack(Wf, Wfp, H, OUTD);
  pack(Wa1, Wa1p, LEN, H3);
  pack(Wa2, Wa2p, H3, LEN);

  // ---- fc stack ----
  ln_to_bf16_kernel<<<M1, TPB, 0, stream>>>(x, ln0_g, ln0_b, xh, FS);
  gemm(xh, W0p, b0, nullptr, hA, nullptr, M1, H, FS, /*relu=*/1);

  float* cur = hA;
  float* nxt = hB;
  for (int i = 0; i < NL; ++i) {
    ln_to_bf16_kernel<<<M1, TPB, 0, stream>>>(cur, rln_g + (size_t)i * H,
                                              rln_b + (size_t)i * H, xh, H);
    gemm(xh, rWp + (size_t)i * H * H, rb + (size_t)i * H,
         /*resid=*/cur, nxt, nullptr, M1, H, H, /*relu=*/1);
    float* t = cur; cur = nxt; nxt = t;
  }
  ln_to_bf16_kernel<<<M1, TPB, 0, stream>>>(cur, lnf_g, lnf_b, xh, H);
  gemm(xh, Wfp, bfb, nullptr, fc, nullptr, M1, OUTD, H, /*relu=*/0);

  // ---- attention over time ----
  mean_transpose_kernel<<<BSZ, TPB, 0, stream>>>(fc, rbuf);
  gemm(rbuf, Wa1p, ba1, nullptr, nullptr, a1, M2, H3, LEN, /*relu=*/1);
  gemm(a1,   Wa2p, ba2, nullptr, wbuf, nullptr, M2, LEN, H3, /*relu=*/0);
  softmax_wsum_kernel<<<M2 / 8, TPB, 0, stream>>>(wbuf, fc, outp);
}
